// VCSMC_858993459240
// MI455X (gfx1250) — matrix operand order repaired and checked by reading, not compiled, not run on hardware
//
#include <hip/hip_runtime.h>
#include <hip/hip_bf16.h>
#include <stdint.h>
#include <math.h>

// ---------------------------------------------------------------------------
// VCSMC forward for MI455X (gfx1250, wave32).
// State resident in d_ws, ping-pong resample/compact, WMMA f16 16x16x32 for
// the encoder GEMM (16x2048x64) and the per-round merge GEMM (128x128x64).
// Fragment loads use contiguous 8xf16 runs (ds_load_b128 / global_load_b128)
// with contiguous k order k = 16*(lane/16) + e, shared by A and B.
// ---------------------------------------------------------------------------

#define NTAXA 16
#define KP    128        // particles
#define SS    512        // sites
#define AA    4          // alphabet
#define DD    64         // embedding dim
#define CC    16         // site-pos encoding dim
#define ROUNDS (NTAXA - 1)
#define PRIOR_BL 1.0f
#define RATEC    1.0f
#define LOG_RATE 0.0f

typedef __attribute__((ext_vector_type(16))) _Float16 v16h;
typedef __attribute__((ext_vector_type(8)))  _Float16 h8;
typedef __attribute__((ext_vector_type(8)))  float    v8f;

// ------------------------------ RNG helpers --------------------------------
__device__ __forceinline__ uint32_t mix32(uint32_t h) {
  h ^= h >> 16; h *= 0x7feb352dU; h ^= h >> 15; h *= 0x846ca68bU; h ^= h >> 16;
  return h;
}
__device__ __forceinline__ uint32_t hash3(uint32_t a, uint32_t b, uint32_t c) {
  return mix32(a * 0x9E3779B1u ^ mix32(b * 0x85EBCA77u ^ mix32(c ^ 0x2545F491u)));
}
__device__ __forceinline__ float u01(uint32_t h) {
  return ((float)((h >> 8) & 0xFFFFFFu) + 0.5f) * (1.0f / 16777216.0f);
}
__device__ __forceinline__ float lse4(const float* t) {
  float m = fmaxf(fmaxf(t[0], t[1]), fmaxf(t[2], t[3]));
  return m + logf(expf(t[0] - m) + expf(t[1] - m) + expf(t[2] - m) + expf(t[3] - m));
}

// ------------------------ WMMA fragment loaders ----------------------------
// A-matrix (f16, 16x32, MxK): lane L holds row M=L&15;
// elements e=0..7  -> K = k0 + koff + e        (koff = 16 if lane>=16)
// elements e=8..15 -> K = k0 + koff + 8 + (e-8)
// i.e. two contiguous 8xf16 runs -> two b128 loads.
__device__ __forceinline__ v16h fragA_lds(const _Float16* A, int lda, int mbase, int k0, int lane) {
  int m = mbase + (lane & 15);
  int koff = (lane & 16) ? 16 : 0;
  h8 lo = *(const h8*)(A + m * lda + k0 + koff);
  h8 hi = *(const h8*)(A + m * lda + k0 + koff + 8);
  v16h out;
#pragma unroll
  for (int e = 0; e < 8; ++e) { out[e] = lo[e]; out[8 + e] = hi[e]; }
  return out;
}
// B-matrix (f16, 32x16, KxN): lane L holds col N=L&15; K = k0 + e + koff
// (koff = 16 if lane>=16).  With B stored TRANSPOSED in LDS as [N][K]
// (ldk = K-stride), the 16 elements are one contiguous run -> two b128 loads.
__device__ __forceinline__ v16h fragB_ldsT(const _Float16* BT, int ldk, int k0, int nbase, int lane) {
  int n = nbase + (lane & 15);
  int koff = (lane & 16) ? 16 : 0;
  h8 lo = *(const h8*)(BT + n * ldk + k0 + koff);
  h8 hi = *(const h8*)(BT + n * ldk + k0 + koff + 8);
  v16h out;
#pragma unroll
  for (int e = 0; e < 8; ++e) { out[e] = lo[e]; out[8 + e] = hi[e]; }
  return out;
}

// --------------------------- init kernels ----------------------------------
// emb0[16,64] = tanh(data[16, 2048] @ W_enc[2048, 64]); one block, 4 waves,
// each wave owns one 16-wide N tile, 64 k-steps of v_wmma_f32_16x16x32_f16.
__global__ __launch_bounds__(128) void k_encoder(const float* __restrict__ data,
                                                 const float* __restrict__ Wenc,
                                                 float* __restrict__ emb0) {
  const int KDIM = SS * AA;  // 2048
  int lane = threadIdx.x & 31;
  int wave = threadIdx.x >> 5;  // = ntile 0..3
  v8f acc = {};
  for (int k0 = 0; k0 < KDIM; k0 += 32) {
    v16h a, b;
    int m = lane & 15;
    int koffA = (lane & 16) ? 16 : 0;
    const float* arow0 = data + m * KDIM + k0 + koffA;       // contiguous 8
    const float* arow1 = data + m * KDIM + k0 + koffA + 8;   // contiguous 8
#pragma unroll
    for (int e = 0; e < 8; ++e) {
      a[e] = (_Float16)arow0[e];
      a[8 + e] = (_Float16)arow1[e];
    }
    int n = wave * 16 + (lane & 15);
    int koffB = (lane & 16) ? 16 : 0;
#pragma unroll
    for (int e = 0; e < 16; ++e) {
      int k = k0 + e + koffB;
      b[e] = (_Float16)Wenc[k * DD + n];
    }
    acc = __builtin_amdgcn_wmma_f32_16x16x32_f16(false, a, false, b, (short)0, acc, false, false);
  }
  int n = wave * 16 + (lane & 15);
#pragma unroll
  for (int i = 0; i < 8; ++i) {
    int m = i + ((lane & 16) ? 8 : 0);
    emb0[m * DD + n] = tanhf(acc[i]);
  }
}

__global__ void k_bcast_emb(const float* __restrict__ emb0, float* __restrict__ E0) {
  int tot = KP * NTAXA * DD;
  for (int i = blockIdx.x * blockDim.x + threadIdx.x; i < tot; i += gridDim.x * blockDim.x)
    E0[i] = emb0[i % (NTAXA * DD)];
}

__global__ void k_init_fel(const float* __restrict__ datab, float* __restrict__ F0) {
  int tot = KP * NTAXA * SS * AA;
  for (int i = blockIdx.x * blockDim.x + threadIdx.x; i < tot; i += gridDim.x * blockDim.x)
    F0[i] = logf(datab[i % (NTAXA * SS * AA)]);
}

__global__ __launch_bounds__(KP) void k_init_misc(float* LOGPI, float* LOGW, int* LC0, float* LOGDF) {
  int k = threadIdx.x;
  LOGPI[k] = 0.f;
  LOGW[k] = 0.f;
  for (int n = 0; n < NTAXA; ++n) LC0[k * NTAXA + n] = 1;
  if (k == 0) {
    LOGDF[0] = 0.f; LOGDF[1] = 0.f;
    for (int i = 2; i <= 2 * NTAXA; ++i) LOGDF[i] = LOGDF[i - 2] + logf((float)i);
  }
}

__global__ void k_sitec(const float* __restrict__ SP, const float* __restrict__ Wpos,
                        float* __restrict__ SITEC) {
  for (int idx = blockIdx.x * blockDim.x + threadIdx.x; idx < SS * CC; idx += gridDim.x * blockDim.x) {
    int c = idx % CC, s = idx / CC;
    float acc = 0.f;
    for (int f = 0; f < SS; ++f) acc += SP[s * SS + f] * Wpos[f * CC + c];
    SITEC[idx] = acc;
  }
}

__global__ void k_sitelogits(const float* __restrict__ SITEC, const float* __restrict__ Wsite,
                             float* __restrict__ SL) {
  for (int idx = blockIdx.x * blockDim.x + threadIdx.x; idx < SS * AA; idx += gridDim.x * blockDim.x) {
    int a = idx % AA, s = idx / AA;
    float acc = 0.f;
    for (int c = 0; c < CC; ++c) acc += SITEC[s * CC + c] * Wsite[c * AA + a];
    SL[idx] = acc;
  }
}

// ------------------------- per-round kernels -------------------------------
// Categorical resample + proposal draws + small-state gather/compact/append.
__global__ __launch_bounds__(KP) void k_resample(
    int r, int t,
    const float* __restrict__ LOGW, const float* __restrict__ LOGPI,
    float* GATHPI, float* LOGLIK, float* LOGVP,
    int* IDXK, int* IDX1, int* IDX2, float* B1, float* B2,
    const int* LCs, int* LCd,
    const int* M1s, int* M1d, const int* M2s, int* M2d,
    const float* BR1s, float* BR1d, const float* BR2s, float* BR2d) {
  __shared__ float w[KP], cdf[KP];
  __shared__ float mx, tot;
  int k = threadIdx.x;
  w[k] = LOGW[k];
  __syncthreads();
  if (k == 0) {
    float m = -1e30f;
    for (int i = 0; i < KP; ++i) m = fmaxf(m, w[i]);
    mx = m;
  }
  __syncthreads();
  w[k] = expf(w[k] - mx);
  __syncthreads();
  if (k == 0) {
    float s = 0.f;
    for (int i = 0; i < KP; ++i) { s += w[i]; cdf[i] = s; }
    tot = s;
  }
  __syncthreads();
  float u = u01(hash3(42u, (uint32_t)(r * 4 + 0), (uint32_t)k)) * tot;
  int sidx = 0;
  while (sidx < KP - 1 && cdf[sidx] < u) sidx++;
  IDXK[k] = sidx;

  int npairs = t * (t - 1) / 2;
  int p = (int)(hash3(42u, (uint32_t)(r * 4 + 1), (uint32_t)k) % (uint32_t)npairs);
  int i1 = 0, rem = p;
  while (rem >= t - 1 - i1) { rem -= t - 1 - i1; i1++; }
  int i2 = i1 + 1 + rem;
  IDX1[k] = i1; IDX2[k] = i2;

  float b1 = -logf(1.0f - u01(hash3(42u, (uint32_t)(r * 4 + 2), (uint32_t)k))) * PRIOR_BL;
  float b2 = -logf(1.0f - u01(hash3(42u, (uint32_t)(r * 4 + 3), (uint32_t)k))) * PRIOR_BL;
  B1[k] = b1; B2[k] = b2;
  LOGVP[k] = -logf((float)npairs) + (LOG_RATE - RATEC * b1) + (LOG_RATE - RATEC * b2);
  GATHPI[k] = LOGPI[sidx];
  LOGLIK[k] = 0.f;

  // leaf counts: gather by sidx, drop i1/i2 preserving order, append merged
  for (int j2 = 0; j2 < t - 2; ++j2) {
    int j = j2 + (j2 >= i1 ? 1 : 0);
    if (j >= i2) j++;
    LCd[k * NTAXA + j2] = LCs[sidx * NTAXA + j];
  }
  LCd[k * NTAXA + (t - 2)] = LCs[sidx * NTAXA + i1] + LCs[sidx * NTAXA + i2];
  // merge / branch histories: gather then append this round's entries
  for (int j = 0; j < r; ++j) {
    M1d[k * ROUNDS + j] = M1s[sidx * ROUNDS + j];
    M2d[k * ROUNDS + j] = M2s[sidx * ROUNDS + j];
    BR1d[k * ROUNDS + j] = BR1s[sidx * ROUNDS + j];
    BR2d[k * ROUNDS + j] = BR2s[sidx * ROUNDS + j];
  }
  M1d[k * ROUNDS + r] = i1;
  M2d[k * ROUNDS + r] = i2;
  BR1d[k * ROUNDS + r] = b1;
  BR2d[k * ROUNDS + r] = b2;
}

// Fused resample-gather + compaction of embeddings, plus [e1|e2] staging.
__global__ void k_gather_emb(int t, const int* __restrict__ IDXK, const int* __restrict__ IDX1,
                             const int* __restrict__ IDX2, const float* __restrict__ Es,
                             float* __restrict__ Ed, float* __restrict__ E1E2) {
  int tot1 = KP * (t - 2) * DD;
  int tot2 = KP * 2 * DD;
  for (int idx = blockIdx.x * blockDim.x + threadIdx.x; idx < tot1 + tot2;
       idx += gridDim.x * blockDim.x) {
    if (idx < tot1) {
      int d = idx % DD, rest = idx / DD;
      int j2 = rest % (t - 2), k = rest / (t - 2);
      int sp = IDXK[k], i1 = IDX1[k], i2 = IDX2[k];
      int j = j2 + (j2 >= i1 ? 1 : 0);
      if (j >= i2) j++;
      Ed[(k * NTAXA + j2) * DD + d] = Es[(sp * NTAXA + j) * DD + d];
    } else {
      int q = idx - tot1;
      int d = q % DD, rest = q / DD;
      int half = rest & 1, k = rest >> 1;
      int sp = IDXK[k];
      int j = half ? IDX2[k] : IDX1[k];
      E1E2[k * (2 * DD) + half * DD + d] = Es[(sp * NTAXA + j) * DD + d];
    }
  }
}

// Fused resample-gather + compaction of log_fel (float4 = one site, A=4),
// plus extraction of the two merged children into F1/F2.
__global__ void k_gather_fel(int t, const int* __restrict__ IDXK, const int* __restrict__ IDX1,
                             const int* __restrict__ IDX2, const float4* __restrict__ Fs,
                             float4* __restrict__ Fd, float4* __restrict__ F1,
                             float4* __restrict__ F2) {
  int tot1 = KP * (t - 2) * SS;
  int tot2 = 2 * KP * SS;
  for (int idx = blockIdx.x * blockDim.x + threadIdx.x; idx < tot1 + tot2;
       idx += gridDim.x * blockDim.x) {
    if (idx < tot1) {
      int s = idx % SS, rest = idx / SS;
      int j2 = rest % (t - 2), k = rest / (t - 2);
      int sp = IDXK[k], i1 = IDX1[k], i2 = IDX2[k];
      int j = j2 + (j2 >= i1 ? 1 : 0);
      if (j >= i2) j++;
      const float4* src = &Fs[(sp * NTAXA + j) * SS + s];
      __builtin_prefetch((const void*)(src + 16), 0, 0);  // global_prefetch_b8
      Fd[(k * NTAXA + j2) * SS + s] = *src;
    } else {
      int q = idx - tot1;
      int s = q % SS, rest = q / SS;
      int half = rest & 1, k = rest >> 1;
      int sp = IDXK[k];
      int j = half ? IDX2[k] : IDX1[k];
      float4 v = Fs[(sp * NTAXA + j) * SS + s];
      if (half) F2[k * SS + s] = v; else F1[k * SS + s] = v;
    }
  }
}

// new_emb[128,64] = tanh([e1|e2][128,128] @ W_merge[128,64]) via WMMA in LDS.
// 8 waves: wave w owns M-tile w, loops 4 N-tiles x 4 k-steps.
// B tile stored transposed ([N][K]) so all fragment loads are ds_load_b128.
__global__ __launch_bounds__(256) void k_merge_gemm(int t, const float* __restrict__ E1E2,
                                                    const float* __restrict__ Wm,
                                                    float* __restrict__ NEWEMB,
                                                    float* __restrict__ Ed) {
  __shared__ _Float16 As[KP * 2 * DD];   // [M=128][K=128], 32 KB
  __shared__ _Float16 BsT[DD * 2 * DD];  // [N=64][K=128], 16 KB (transposed)
  int tid = threadIdx.x;
  for (int i = tid; i < KP * 2 * DD; i += 256) As[i] = (_Float16)E1E2[i];
  for (int i = tid; i < 2 * DD * DD; i += 256) {
    int k = i / DD, n = i % DD;                 // Wm is [K=128][N=64] row-major
    BsT[n * (2 * DD) + k] = (_Float16)Wm[i];
  }
  __syncthreads();
  int wave = tid >> 5, lane = tid & 31;
  int mbase = wave * 16;
  for (int ntile = 0; ntile < 4; ++ntile) {
    v8f acc = {};
#pragma unroll
    for (int kk = 0; kk < 4; ++kk) {
      int k0 = kk * 32;
      v16h a = fragA_lds(As, 2 * DD, mbase, k0, lane);
      v16h b = fragB_ldsT(BsT, 2 * DD, k0, ntile * 16, lane);
      acc = __builtin_amdgcn_wmma_f32_16x16x32_f16(false, a, false, b, (short)0, acc, false, false);
    }
    int n = ntile * 16 + (lane & 15);
#pragma unroll
    for (int i = 0; i < 8; ++i) {
      int m = mbase + i + ((lane & 16) ? 8 : 0);  // m == particle index
      float v = tanhf(acc[i]);
      NEWEMB[m * DD + n] = v;
      Ed[(m * NTAXA + (t - 2)) * DD + n] = v;     // append merged row at slot t-2
    }
  }
}

// Felsenstein pruning at the new node: F81 closed-form P(t), fel-half via
// logsumexp over A=4 for both branches; writes log_fel slot t-2.
__global__ __launch_bounds__(256) void k_newfel(int t, const float* __restrict__ NEWEMB,
                                                const float* __restrict__ Wstat,
                                                const float* __restrict__ SL,
                                                const float* __restrict__ B1,
                                                const float* __restrict__ B2,
                                                const float* __restrict__ F1,
                                                const float* __restrict__ F2,
                                                float* __restrict__ Fd) {
  int k = blockIdx.x;
  __shared__ float base[AA];
  if (threadIdx.x < AA) {
    float s = 0.f;
    for (int d = 0; d < DD; ++d) s += NEWEMB[k * DD + d] * Wstat[d * AA + threadIdx.x];
    base[threadIdx.x] = s;
  }
  __syncthreads();
  float e1 = expf(-B1[k]), e2 = expf(-B2[k]);
  for (int s = threadIdx.x; s < SS; s += blockDim.x) {
    float l[AA];
    float m = -1e30f;
#pragma unroll
    for (int a = 0; a < AA; ++a) { l[a] = base[a] + SL[s * AA + a]; m = fmaxf(m, l[a]); }
    float Z = 0.f;
#pragma unroll
    for (int a = 0; a < AA; ++a) { l[a] = expf(l[a] - m); Z += l[a]; }
    float pi[AA];
#pragma unroll
    for (int a = 0; a < AA; ++a) pi[a] = l[a] / Z;
    float f1v[AA], f2v[AA];
#pragma unroll
    for (int b = 0; b < AA; ++b) {
      f1v[b] = F1[(k * SS + s) * AA + b];
      f2v[b] = F2[(k * SS + s) * AA + b];
    }
#pragma unroll
    for (int a = 0; a < AA; ++a) {
      float t1[AA], t2[AA];
#pragma unroll
      for (int b = 0; b < AA; ++b) {
        float P1 = (1.f - e1) * pi[b] + ((a == b) ? e1 : 0.f);
        float P2 = (1.f - e2) * pi[b] + ((a == b) ? e2 : 0.f);
        t1[b] = logf(P1 + 1e-30f) + f1v[b];
        t2[b] = logf(P2 + 1e-30f) + f2v[b];
      }
      Fd[((k * NTAXA + (t - 2)) * SS + s) * AA + a] = lse4(t1) + lse4(t2);
    }
  }
}

// Forest log-likelihood: sum over nodes & sites of logsumexp_a(fel + log_stat).
__global__ __launch_bounds__(256) void k_loglik(int t, const float* __restrict__ Ed,
                                                const float* __restrict__ Wstat,
                                                const float* __restrict__ SL,
                                                const float* __restrict__ Fd,
                                                float* __restrict__ LOGLIK) {
  int tn = t - 1;
  int node = blockIdx.x % tn;
  int k = blockIdx.x / tn;
  __shared__ float base[AA];
  if (threadIdx.x < AA) {
    float s = 0.f;
    for (int d = 0; d < DD; ++d) s += Ed[(k * NTAXA + node) * DD + d] * Wstat[d * AA + threadIdx.x];
    base[threadIdx.x] = s;
  }
  __syncthreads();
  float acc = 0.f;
  for (int s = threadIdx.x; s < SS; s += blockDim.x) {
    float l[AA];
    float m = -1e30f;
#pragma unroll
    for (int a = 0; a < AA; ++a) { l[a] = base[a] + SL[s * AA + a]; m = fmaxf(m, l[a]); }
    float Z = 0.f;
#pragma unroll
    for (int a = 0; a < AA; ++a) Z += expf(l[a] - m);
    float lZ = m + logf(Z);
    float terms[AA];
#pragma unroll
    for (int a = 0; a < AA; ++a)
      terms[a] = Fd[((k * NTAXA + node) * SS + s) * AA + a] + (l[a] - lZ);
    acc += lse4(terms);
  }
  __shared__ float red[256];
  red[threadIdx.x] = acc;
  __syncthreads();
  for (int off = 128; off > 0; off >>= 1) {
    if (threadIdx.x < off) red[threadIdx.x] += red[threadIdx.x + off];
    __syncthreads();
  }
  if (threadIdx.x == 0) atomicAdd(&LOGLIK[k], red[0]);
}

__global__ __launch_bounds__(KP) void k_weights(int r, int t, const float* __restrict__ LOGLIK,
                                                const int* __restrict__ LCd,
                                                const float* __restrict__ BR1d,
                                                const float* __restrict__ BR2d,
                                                const float* __restrict__ GATHPI,
                                                const float* __restrict__ LOGVP,
                                                const float* __restrict__ LOGDF, float* LOGPI,
                                                float* LOGW, float* LOGWS) {
  int k = threadIdx.x;
  float lb = 0.f;
  for (int j = 0; j <= r; ++j)
    lb += 2.f * LOG_RATE - RATEC * (BR1d[k * ROUNDS + j] + BR2d[k * ROUNDS + j]);
  float lt = 0.f;
  for (int j = 0; j < t - 1; ++j) {
    int lc = LCd[k * NTAXA + j];
    int ii = 2 * lc - 3;
    if (ii < 0) ii = 0;
    lt -= LOGDF[ii];
  }
  float npi = LOGLIK[k] + lt + lb;
  float lw = npi - GATHPI[k] - LOGVP[k];
  LOGPI[k] = npi;
  LOGW[k] = lw;
  LOGWS[r * KP + k] = lw;
}

__global__ __launch_bounds__(KP) void k_finalize(const float* __restrict__ LOGWS,
                                                 const float* __restrict__ LOGLIK,
                                                 const int* __restrict__ M1,
                                                 const int* __restrict__ M2,
                                                 const float* __restrict__ BR1,
                                                 const float* __restrict__ BR2,
                                                 float* __restrict__ out) {
  __shared__ float sh[KP];
  __shared__ float logZ;
  __shared__ int best;
  int k = threadIdx.x;
  if (k == 0) logZ = 0.f;
  __syncthreads();
  for (int r = 0; r < ROUNDS; ++r) {
    sh[k] = LOGWS[r * KP + k];
    __syncthreads();
    if (k == 0) {
      float m = -1e30f;
      for (int i = 0; i < KP; ++i) m = fmaxf(m, sh[i]);
      float s = 0.f;
      for (int i = 0; i < KP; ++i) s += expf(sh[i] - m);
      logZ += m + logf(s) - logf((float)KP);
    }
    __syncthreads();
  }
  if (k == 0) {
    float m = -1e30f;
    int bi = 0;
    for (int i = 0; i < KP; ++i)
      if (LOGLIK[i] > m) { m = LOGLIK[i]; bi = i; }
    best = bi;
    out[0] = logZ;
  }
  __syncthreads();
  out[1 + k] = LOGLIK[k];
  if (k < ROUNDS) {
    out[1 + KP + k] = (float)M1[best * ROUNDS + k];
    out[1 + KP + ROUNDS + k] = (float)M2[best * ROUNDS + k];
    out[1 + KP + 2 * ROUNDS + k] = BR1[best * ROUNDS + k];
    out[1 + KP + 3 * ROUNDS + k] = BR2[best * ROUNDS + k];
  }
}

// ------------------------------- host side ---------------------------------
extern "C" void kernel_launch(void* const* d_in, const int* in_sizes, int n_in, void* d_out,
                              int out_size, void* d_ws, size_t ws_size, hipStream_t stream) {
  const float* data = (const float*)d_in[0];      // 16x512x4
  const float* datab = (const float*)d_in[1];     // 16x512x4
  const float* sitepos = (const float*)d_in[2];   // 512x512
  const float* Wenc = (const float*)d_in[3];      // 2048x64
  const float* Wm = (const float*)d_in[4];        // 128x64
  const float* Wstat = (const float*)d_in[5];     // 64x4
  const float* Wpos = (const float*)d_in[6];      // 512x16
  const float* Wsite = (const float*)d_in[7];     // 16x4
  float* out = (float*)d_out;

  char* base = (char*)d_ws;
  size_t off = 0;
  auto alloc = [&](size_t bytes) -> void* {
    void* r = base + off;
    off = (off + bytes + 255) & ~(size_t)255;
    return r;
  };
  // ping-pong state (~37 MB total; d_ws assumed large enough)
  float* EMB[2] = {(float*)alloc(KP * NTAXA * DD * 4), (float*)alloc(KP * NTAXA * DD * 4)};
  float* FEL[2] = {(float*)alloc((size_t)KP * NTAXA * SS * AA * 4),
                   (float*)alloc((size_t)KP * NTAXA * SS * AA * 4)};
  int* LC[2] = {(int*)alloc(KP * NTAXA * 4), (int*)alloc(KP * NTAXA * 4)};
  int* M1[2] = {(int*)alloc(KP * ROUNDS * 4), (int*)alloc(KP * ROUNDS * 4)};
  int* M2[2] = {(int*)alloc(KP * ROUNDS * 4), (int*)alloc(KP * ROUNDS * 4)};
  float* BR1[2] = {(float*)alloc(KP * ROUNDS * 4), (float*)alloc(KP * ROUNDS * 4)};
  float* BR2[2] = {(float*)alloc(KP * ROUNDS * 4), (float*)alloc(KP * ROUNDS * 4)};
  float* F1 = (float*)alloc(KP * SS * AA * 4);
  float* F2 = (float*)alloc(KP * SS * AA * 4);
  float* E1E2 = (float*)alloc(KP * 2 * DD * 4);
  float* NEWEMB = (float*)alloc(KP * DD * 4);
  float* LOGPI = (float*)alloc(KP * 4);
  float* LOGW = (float*)alloc(KP * 4);
  float* GATHPI = (float*)alloc(KP * 4);
  float* LOGLIK = (float*)alloc(KP * 4);
  float* LOGVP = (float*)alloc(KP * 4);
  float* B1 = (float*)alloc(KP * 4);
  float* B2 = (float*)alloc(KP * 4);
  int* IDXK = (int*)alloc(KP * 4);
  int* IDX1 = (int*)alloc(KP * 4);
  int* IDX2 = (int*)alloc(KP * 4);
  float* LOGWS = (float*)alloc(ROUNDS * KP * 4);
  float* SITEC = (float*)alloc(SS * CC * 4);
  float* SL = (float*)alloc(SS * AA * 4);
  float* LOGDF = (float*)alloc(64 * 4);
  float* EMB0V = (float*)alloc(NTAXA * DD * 4);
  (void)in_sizes; (void)n_in; (void)out_size; (void)ws_size;

  // ---- init ----
  k_encoder<<<1, 128, 0, stream>>>(data, Wenc, EMB0V);
  k_bcast_emb<<<512, 256, 0, stream>>>(EMB0V, EMB[0]);
  k_init_fel<<<4096, 256, 0, stream>>>(datab, FEL[0]);
  k_init_misc<<<1, KP, 0, stream>>>(LOGPI, LOGW, LC[0], LOGDF);
  k_sitec<<<32, 256, 0, stream>>>(sitepos, Wpos, SITEC);
  k_sitelogits<<<8, 256, 0, stream>>>(SITEC, Wsite, SL);

  // ---- SMC rounds ----
  for (int r = 0; r < ROUNDS; ++r) {
    int t = NTAXA - r;
    int src = r & 1, dst = src ^ 1;
    k_resample<<<1, KP, 0, stream>>>(r, t, LOGW, LOGPI, GATHPI, LOGLIK, LOGVP, IDXK, IDX1, IDX2,
                                     B1, B2, LC[src], LC[dst], M1[src], M1[dst], M2[src], M2[dst],
                                     BR1[src], BR1[dst], BR2[src], BR2[dst]);
    k_gather_emb<<<256, 256, 0, stream>>>(t, IDXK, IDX1, IDX2, EMB[src], EMB[dst], E1E2);
    k_gather_fel<<<4096, 256, 0, stream>>>(t, IDXK, IDX1, IDX2, (const float4*)FEL[src],
                                           (float4*)FEL[dst], (float4*)F1, (float4*)F2);
    k_merge_gemm<<<1, 256, 0, stream>>>(t, E1E2, Wm, NEWEMB, EMB[dst]);
    k_newfel<<<KP, 256, 0, stream>>>(t, NEWEMB, Wstat, SL, B1, B2, F1, F2, FEL[dst]);
    k_loglik<<<KP * (t - 1), 256, 0, stream>>>(t, EMB[dst], Wstat, SL, FEL[dst], LOGLIK);
    k_weights<<<1, KP, 0, stream>>>(r, t, LOGLIK, LC[dst], BR1[dst], BR2[dst], GATHPI, LOGVP,
                                    LOGDF, LOGPI, LOGW, LOGWS);
  }

  // final state lives in buffer ((ROUNDS-1)&1)^1 == 1
  k_finalize<<<1, KP, 0, stream>>>(LOGWS, LOGLIK, M1[1], M2[1], BR1[1], BR2[1], out);
}
